// SyncLoss_24215025615023
// MI455X (gfx1250) — matrix operand order repaired and checked by reading, not compiled, not run on hardware
//
#include <hip/hip_runtime.h>
#include <hip/hip_bf16.h>
#include <math.h>

typedef __attribute__((ext_vector_type(16))) _Float16 v16h;
typedef __attribute__((ext_vector_type(8)))  _Float16 v8h;
typedef __attribute__((ext_vector_type(8)))  float    v8f;

#define NB   64      // batches
#define NF   512     // features (K of the GEMM)
#define NT   600     // time steps
#define NM   200     // merged segments
#define MP   208     // padded M (13 * 16)
#define MT   13      // M tiles of 16
#define KC   32      // K chunks of 16 features
#define NTPW 4       // ntiles per wave in the GEMM
#define NTG  4       // ntile groups (last group overlaps: starts at MT-NTPW)
#define EPSV 1e-8f

// t indices contributing to segment m
__device__ __forceinline__ void seg_ts(int m, int& t0, int& t1, int& t2) {
    if (m < 199) { t0 = 3 * m + 1; t1 = 3 * m + 2; t2 = 3 * m + 3; }
    else         { t0 = 0;         t1 = 598;       t2 = 599;       }
}

// ---------------------------------------------------------------------------
// Stage 1: merge over T and write f16 operand blobs in WMMA-swizzled layout.
// grid = (64, 32, 2), block = 256.  z==0: feature_a -> A blob, z==1: feature_b -> B blob
// A blob (halfs): ((((b*32 + kc)*13 + mtile)*32 + lane)*8 + e)   k_local = (lane/16)*8 + e
// B blob (halfs): ((((b*32 + kc)*13 + ntile)*16 + n)*16 + kl)    row n, k_local = kl
// ---------------------------------------------------------------------------
__global__ void merge_swizzle_kernel(const float* __restrict__ fa,
                                     const float* __restrict__ fb,
                                     _Float16* __restrict__ ablob,
                                     _Float16* __restrict__ bblob) {
    const int b  = blockIdx.x;
    const int kc = blockIdx.y;
    const int z  = blockIdx.z;
    const int tid = threadIdx.x;

    __shared__ float tile[16 * NT];  // 16 feature rows x 600 timesteps = 38.4 KB

    const float* src = (z == 0 ? fa : fb) + ((size_t)b * NF + (size_t)kc * 16) * NT;
    for (int i = tid; i < 16 * NT; i += 256) tile[i] = src[i];
    __syncthreads();

    if (z == 0) {
        // A-operand blob
        const int lane = tid >> 3;          // 0..31
        const int e    = tid & 7;           // 0..7
        const int mloc = lane & 15;
        const int fl   = ((lane >> 4) << 3) + e;   // 0..15
        const float* row = &tile[fl * NT];
        for (int mtile = 0; mtile < MT; ++mtile) {
            const int m = mtile * 16 + mloc;
            float v = 0.0f;
            if (m < NM) {
                int t0, t1, t2; seg_ts(m, t0, t1, t2);
                v = row[t0] + row[t1] + row[t2];
            }
            size_t idx = ((((size_t)b * KC + kc) * MT + mtile) * 32 + lane) * 8 + e;
            ablob[idx] = (_Float16)v;
        }
    } else {
        // B-operand blob
        const int n  = tid >> 4;            // 0..15
        const int kl = tid & 15;            // 0..15
        const float* row = &tile[kl * NT];
        for (int ntile = 0; ntile < MT; ++ntile) {
            const int m = ntile * 16 + n;
            float v = 0.0f;
            if (m < NM) {
                int t0, t1, t2; seg_ts(m, t0, t1, t2);
                v = row[t0] + row[t1] + row[t2];
            }
            size_t idx = ((((size_t)b * KC + kc) * MT + ntile) * 16 + n) * 16 + kl;
            bblob[idx] = (_Float16)v;
        }
    }
}

// ---------------------------------------------------------------------------
// Stage 2: squared norms from the f16 blobs (deterministic, no atomics).
// grid = (64, 208, 2), block = 32 (one wave per (b, m)).
// ---------------------------------------------------------------------------
__global__ void norms_kernel(const _Float16* __restrict__ ablob,
                             const _Float16* __restrict__ bblob,
                             float* __restrict__ norm2a,
                             float* __restrict__ norm2b) {
    const int b = blockIdx.x;
    const int m = blockIdx.y;
    const int z = blockIdx.z;
    const int lane = threadIdx.x;

    float acc = 0.0f;
    if (z == 0) {
        const int mtile = m >> 4, mloc = m & 15;
        for (int f = lane; f < NF; f += 32) {
            const int kc = f >> 4, fl = f & 15;
            const int l  = ((fl >> 3) << 4) + mloc;
            const int e  = fl & 7;
            size_t idx = ((((size_t)b * KC + kc) * MT + mtile) * 32 + l) * 8 + e;
            float v = (float)ablob[idx];
            acc += v * v;
        }
    } else {
        const int ntile = m >> 4, n = m & 15;
        for (int f = lane; f < NF; f += 32) {
            const int kc = f >> 4, kl = f & 15;
            size_t idx = ((((size_t)b * KC + kc) * MT + ntile) * 16 + n) * 16 + kl;
            float v = (float)bblob[idx];
            acc += v * v;
        }
    }
    for (int off = 16; off > 0; off >>= 1) acc += __shfl_xor(acc, off, 32);
    if (lane == 0) {
        float* dst = (z == 0 ? norm2a : norm2b);
        dst[b * MP + m] = acc;
    }
}

// ---------------------------------------------------------------------------
// Stage 3: batched GEMM via v_wmma_f32_16x16x32_f16 + cosine epilogue.
// grid = (64, 13, 4), block = 32.  One wave computes 1 mtile x exactly-4 ntiles
// (last group overlaps: tiles 9..12), reusing the A operand across the 4 WMMAs
// of each K-step.  No runtime tile guard -> flat unroll, no spills.
// ---------------------------------------------------------------------------
__global__ void gemm_wmma_kernel(const _Float16* __restrict__ ablob,
                                 const _Float16* __restrict__ bblob,
                                 const float* __restrict__ norm2a,
                                 const float* __restrict__ norm2b,
                                 float* __restrict__ cosbuf) {
    const int b     = blockIdx.x;
    const int mtile = blockIdx.y;
    const int ng    = blockIdx.z;        // ntile group
    const int lane  = threadIdx.x;       // 0..31
    const int g     = lane >> 4;         // lane group
    const int nloc  = lane & 15;

    // groups start at 0, 4, 8, 9 -> every group covers exactly NTPW valid tiles;
    // tiles 9..11 are computed twice with bitwise-identical results (benign).
    const int nt0 = (ng == NTG - 1) ? (MT - NTPW) : ng * NTPW;

    v8f c[NTPW];
    #pragma unroll
    for (int j = 0; j < NTPW; ++j) c[j] = (v8f){};

    for (int kb = 0; kb < 16; ++kb) {    // 16 * K32 = 512
        // A: two 16-byte chunks (k 0..7 and 8..15 of the lane group's K16 chunk)
        const int c0 = 2 * kb + g, c1 = 2 * kb + g;
        const v8h* pa0 = (const v8h*)&ablob[((((size_t)b * KC + c0) * MT + mtile) * 32 + nloc) * 8];
        const v8h* pa1 = (const v8h*)&ablob[((((size_t)b * KC + c1) * MT + mtile) * 32 + 16 + nloc) * 8];
        union { v16h v; v8h h[2]; } ua;
        ua.h[0] = *pa0;
        ua.h[1] = *pa1;
        // B: lane group selects which K16 chunk; 32 contiguous bytes per lane
        const size_t bbase = (((size_t)b * KC + (2 * kb + g)) * MT) * 256 + (size_t)nloc * 16;
        #pragma unroll
        for (int j = 0; j < NTPW; ++j) {
            const v16h* pb = (const v16h*)&bblob[bbase + (size_t)(nt0 + j) * 256];
            v16h bv = *pb;
            c[j] = __builtin_amdgcn_wmma_f32_16x16x32_f16(
                    /*neg_a=*/false, ua.v, /*neg_b=*/false, bv,
                    /*c_mod=*/(short)0, c[j], /*reuse_a=*/false, /*reuse_b=*/false);
        }
    }

    // Epilogue: cos = num / max(sqrt(na2*nb2), eps)
    float* out = cosbuf + (size_t)b * MP * MP;
    #pragma unroll
    for (int j = 0; j < NTPW; ++j) {
        const int n = (nt0 + j) * 16 + nloc;
        const float nb2 = norm2b[b * MP + n];
        #pragma unroll
        for (int r = 0; r < 8; ++r) {
            const int m = mtile * 16 + r + 8 * g;
            const float na2 = norm2a[b * MP + m];
            const float denom = fmaxf(sqrtf(na2 * nb2), EPSV);
            out[(size_t)m * MP + n] = c[j][r] / denom;
        }
    }
}

// ---------------------------------------------------------------------------
// Stage 4: per-row -logp(diag)/200.  grid = (64, 200), block = 32.
// ---------------------------------------------------------------------------
__global__ void rowloss_kernel(const float* __restrict__ cosbuf,
                               float* __restrict__ rowloss) {
    const int b = blockIdx.x;
    const int m = blockIdx.y;
    const int lane = threadIdx.x;
    const float* row = cosbuf + ((size_t)b * MP + m) * MP;

    float vals[7];
    int cnt = 0;
    float maxv = -INFINITY;
    for (int n = lane; n < NM; n += 32) {
        float v = row[n];
        vals[cnt++] = v;
        maxv = fmaxf(maxv, v);
    }
    for (int off = 16; off > 0; off >>= 1) maxv = fmaxf(maxv, __shfl_xor(maxv, off, 32));

    float s = 0.0f;
    for (int i = 0; i < cnt; ++i) s += __expf(vals[i] - maxv);
    for (int off = 16; off > 0; off >>= 1) s += __shfl_xor(s, off, 32);

    if (lane == 0) {
        const float diag = row[m];
        const float logp = (diag - maxv) - __logf(s);
        rowloss[b * NM + m] = -logp * (1.0f / (float)NM);
    }
}

// ---------------------------------------------------------------------------
// Stage 5: deterministic final reduction (fixed order).  1 block of 256.
// ---------------------------------------------------------------------------
__global__ void reduce_final_kernel(const float* __restrict__ rowloss,
                                    float* __restrict__ out) {
    __shared__ float red[256];
    const int tid = threadIdx.x;
    float acc = 0.0f;
    for (int i = tid; i < NB * NM; i += 256) acc += rowloss[i];
    red[tid] = acc;
    __syncthreads();
    for (int off = 128; off > 0; off >>= 1) {
        if (tid < off) red[tid] += red[tid + off];
        __syncthreads();
    }
    if (tid == 0) out[0] = red[0];
}

// ---------------------------------------------------------------------------
extern "C" void kernel_launch(void* const* d_in, const int* in_sizes, int n_in,
                              void* d_out, int out_size, void* d_ws, size_t ws_size,
                              hipStream_t stream) {
    // inputs: [0] data_label (int64, UNUSED), [1] feature_a (f32), [2] feature_b (f32)
    const float* fa = (const float*)d_in[1];
    const float* fb = (const float*)d_in[2];
    float* out = (float*)d_out;

    // workspace carve-up (256B aligned)
    char* p = (char*)d_ws;
    auto take = [&](size_t bytes) -> char* {
        char* r = p;
        p += (bytes + 255) & ~(size_t)255;
        return r;
    };
    const size_t nBlobHalf = (size_t)NB * KC * MT * 256;       // 6,815,744 halfs
    _Float16* ablob  = (_Float16*)take(nBlobHalf * sizeof(_Float16));
    _Float16* bblob  = (_Float16*)take(nBlobHalf * sizeof(_Float16));
    float*    norm2a = (float*)take((size_t)NB * MP * sizeof(float));
    float*    norm2b = (float*)take((size_t)NB * MP * sizeof(float));
    float*    cosbuf = (float*)take((size_t)NB * MP * MP * sizeof(float));
    float*    rloss  = (float*)take((size_t)NB * NM * sizeof(float));

    merge_swizzle_kernel<<<dim3(NB, KC, 2), 256, 0, stream>>>(fa, fb, ablob, bblob);
    norms_kernel<<<dim3(NB, MP, 2), 32, 0, stream>>>(ablob, bblob, norm2a, norm2b);
    gemm_wmma_kernel<<<dim3(NB, MT, NTG), 32, 0, stream>>>(ablob, bblob, norm2a, norm2b, cosbuf);
    rowloss_kernel<<<dim3(NB, NM), 32, 0, stream>>>(cosbuf, rloss);
    reduce_final_kernel<<<1, 256, 0, stream>>>(rloss, out);
}
